// MultiHeadedAttention_17351667876501
// MI455X (gfx1250) — matrix operand order repaired and checked by reading, not compiled, not run on hardware
//
#include <hip/hip_runtime.h>

typedef _Float16 f16;
typedef __attribute__((ext_vector_type(16))) _Float16 v16h;
typedef __attribute__((ext_vector_type(8)))  _Float16 v8h;
typedef __attribute__((ext_vector_type(8)))  float    v8f;
typedef __attribute__((ext_vector_type(4)))  float    v4f;
typedef __attribute__((ext_vector_type(4)))  int      v4i;

#define NB   4
#define SEQ  2048
#define BDIM 1024
#define NH   16
#define DKD  64

// ---- CDNA5 async global->LDS copy (ASYNCcnt path), guarded by __has_builtin.
// Probe-learned prototype: (v4i addrspace(1)*, v4i addrspace(3)*, imm offset, imm cpol)
#if __has_builtin(__builtin_amdgcn_global_load_async_to_lds_b128)
#define HAVE_ASYNC_LDS 1
typedef __attribute__((address_space(1))) v4i v4i_gl;
typedef __attribute__((address_space(3))) v4i v4i_lds;
static __device__ __forceinline__ void async_cp32B(f16* d, const f16* s) {
    __builtin_amdgcn_global_load_async_to_lds_b128((v4i_gl*)s,       (v4i_lds*)d,       0, 0);
    __builtin_amdgcn_global_load_async_to_lds_b128((v4i_gl*)(s + 8), (v4i_lds*)(d + 8), 0, 0);
}
static __device__ __forceinline__ void wait_async0() {
#if __has_builtin(__builtin_amdgcn_s_wait_asynccnt)
    __builtin_amdgcn_s_wait_asynccnt(0);
#else
    asm volatile("s_wait_asynccnt 0x0" ::: "memory");
#endif
}
#endif

static __device__ __forceinline__ v16h cat8(v8h lo, v8h hh) {
    v16h r;
    #pragma unroll
    for (int j = 0; j < 8; ++j) { r[j] = lo[j]; r[j + 8] = hh[j]; }
    return r;
}
static __device__ __forceinline__ v8f zero8() {
    v8f z;
    #pragma unroll
    for (int j = 0; j < 8; ++j) z[j] = 0.0f;
    return z;
}
static __device__ __forceinline__ v8f wmma_f16(v16h a, v16h b, v8f c) {
    return __builtin_amdgcn_wmma_f32_16x16x32_f16(
        false, a, false, b, (short)0, c, false, false);
}

// A-fragment (16x32, MxK): lane m = lane&15, hi = lane>>4.
// element j: k = (j<8?0:16) + 8*hi + (j&7)
static __device__ __forceinline__ v16h frag_a(const f16* tile, int ld, int m, int hi) {
    const f16* p = tile + m * ld;
    return cat8(*(const v8h*)(p + 8 * hi), *(const v8h*)(p + 16 + 8 * hi));
}
// B-fragment (32x16, KxN) from LDS stored [n][k]: lane n = lane&15,
// element j: k = (j<8?0:16) + 8*hi + (j&7)
static __device__ __forceinline__ v16h frag_b(const f16* tile, int ld, int n, int hi) {
    const f16* p = tile + n * ld;
    return cat8(*(const v8h*)(p + 8 * hi), *(const v8h*)(p + 16 + 8 * hi));
}
// fp32x16 -> two v8h
static __device__ __forceinline__ void cvt16(const float* s, v8h& h0, v8h& h1) {
    v4f x0 = *(const v4f*)(s);
    v4f x1 = *(const v4f*)(s + 4);
    v4f x2 = *(const v4f*)(s + 8);
    v4f x3 = *(const v4f*)(s + 12);
    #pragma unroll
    for (int j = 0; j < 4; ++j) {
        h0[j] = (f16)x0[j]; h0[j + 4] = (f16)x1[j];
        h1[j] = (f16)x2[j]; h1[j + 4] = (f16)x3[j];
    }
}

// ---------------------------------------------------------------------------
// Kernel 1: fused QKV projection.  Y = X @ W^T + b, fp32 in -> f16 out in
// [B,H,S,DK] layout.  blockIdx.z selects {Q,K,V}.  Tile 128x128, K-step 32.
// ---------------------------------------------------------------------------
__global__ __launch_bounds__(256)
void qkv_proj_kernel(const float* __restrict__ xq, const float* __restrict__ xk,
                     const float* __restrict__ xv,
                     const float* __restrict__ Wq, const float* __restrict__ bq,
                     const float* __restrict__ Wk, const float* __restrict__ bk,
                     const float* __restrict__ Wv, const float* __restrict__ bv,
                     f16* __restrict__ Q16, f16* __restrict__ K16, f16* __restrict__ V16)
{
    const int z = blockIdx.z;
    const float* X    = (z == 0) ? xq : (z == 1) ? xk : xv;
    const float* W    = (z == 0) ? Wq : (z == 1) ? Wk : Wv;
    const float* bias = (z == 0) ? bq : (z == 1) ? bk : bv;
    f16*         dst  = (z == 0) ? Q16 : (z == 1) ? K16 : V16;

    __shared__ f16 lA[128 * 40];
    __shared__ f16 lB[128 * 40];

    const int t    = threadIdx.x;
    const int lane = t & 31;
    const int w    = t >> 5;
    const int hi   = lane >> 4;
    const int ln   = lane & 15;
    const int wm   = (w & 3) * 32;
    const int wn   = (w >> 2) * 64;

    const int Mbase = blockIdx.y * 128;
    const int Nbase = blockIdx.x * 128;
    const int fr = t >> 1;
    const int fc = (t & 1) * 16;

    v8f acc[2][4];
    #pragma unroll
    for (int i = 0; i < 2; ++i)
        #pragma unroll
        for (int j = 0; j < 4; ++j) acc[i][j] = zero8();

    for (int k0 = 0; k0 < BDIM; k0 += 32) {
        {   // A tile: X rows, fp32 -> f16
            v8h h0, h1;
            cvt16(X + (size_t)(Mbase + fr) * BDIM + k0 + fc, h0, h1);
            *(v8h*)(&lA[fr * 40 + fc])     = h0;
            *(v8h*)(&lA[fr * 40 + fc + 8]) = h1;
        }
        {   // B tile: W rows (NT gemm -> contiguous along k)
            v8h h0, h1;
            cvt16(W + (size_t)(Nbase + fr) * BDIM + k0 + fc, h0, h1);
            *(v8h*)(&lB[fr * 40 + fc])     = h0;
            *(v8h*)(&lB[fr * 40 + fc + 8]) = h1;
        }
        __syncthreads();

        v16h af[2], bf[4];
        #pragma unroll
        for (int mt = 0; mt < 2; ++mt)
            af[mt] = frag_a(&lA[(wm + mt * 16) * 40], 40, ln, hi);
        #pragma unroll
        for (int nt = 0; nt < 4; ++nt)
            bf[nt] = frag_b(&lB[(wn + nt * 16) * 40], 40, ln, hi);
        #pragma unroll
        for (int mt = 0; mt < 2; ++mt)
            #pragma unroll
            for (int nt = 0; nt < 4; ++nt)
                acc[mt][nt] = wmma_f16(af[mt], bf[nt], acc[mt][nt]);
        __syncthreads();
    }

    // Epilogue: bias add, f16 scatter into [B,H,S,DK]
    #pragma unroll
    for (int nt = 0; nt < 4; ++nt) {
        const int e  = Nbase + wn + nt * 16 + ln;
        const float be = bias[e];
        const int h  = e >> 6;
        const int dk = e & 63;
        #pragma unroll
        for (int mt = 0; mt < 2; ++mt) {
            #pragma unroll
            for (int r = 0; r < 8; ++r) {
                const int i = Mbase + wm + mt * 16 + r + 8 * hi;
                const int b = i >> 11;
                const int s = i & 2047;
                dst[(((size_t)(b * NH + h)) * SEQ + s) * DKD + dk] =
                    (f16)(acc[mt][nt][r] + be);
            }
        }
    }
}

// ---------------------------------------------------------------------------
// Kernel 2: flash attention.  One WG per (b, h, 128 q rows); wave = 16 q rows.
// KV chunk = 64 staged in LDS (K via async-to-LDS, V transposed at store).
// ---------------------------------------------------------------------------
__global__ __launch_bounds__(256)
void attention_kernel(const f16* __restrict__ Q16, const f16* __restrict__ K16,
                      const f16* __restrict__ V16, const int* __restrict__ mask,
                      f16* __restrict__ O16)
{
    __shared__ f16 Kl[64 * 72];        // [kv][dk]
    __shared__ f16 Vt[64 * 72];        // [dk][kv] transposed
    __shared__ f16 Pl[8 * 16 * 72];    // per-wave P staging

    const int t    = threadIdx.x;
    const int lane = t & 31;
    const int w    = t >> 5;
    const int hi   = lane >> 4;
    const int ln   = lane & 15;

    const int b = blockIdx.z, h = blockIdx.y, qblk = blockIdx.x;
    const size_t hoff = ((size_t)(b * NH + h)) * SEQ * DKD;
    const f16* Qh = Q16 + hoff;
    const f16* Kh = K16 + hoff;
    const f16* Vh = V16 + hoff;
    const int  qb = qblk * 128 + w * 16;

    // Resident Q fragments (DK=64 -> 2 k-steps of 32)
    v16h qf[2];
    #pragma unroll
    for (int ks = 0; ks < 2; ++ks) {
        const f16* p = Qh + (size_t)(qb + ln) * DKD + ks * 32;
        qf[ks] = cat8(*(const v8h*)(p + 8 * hi), *(const v8h*)(p + 16 + 8 * hi));
    }

    v8f o[4];
    #pragma unroll
    for (int nt = 0; nt < 4; ++nt) o[nt] = zero8();
    float M[8], L[8];
    #pragma unroll
    for (int r = 0; r < 8; ++r) { M[r] = -3.0e30f; L[r] = 0.0f; }

    f16* Pw = &Pl[w * 16 * 72];

    for (int kv0 = 0; kv0 < SEQ; kv0 += 64) {
        {   // K chunk -> LDS, row-major (async path when available)
            const int j = t >> 2, cb = (t & 3) * 16;
            const f16* s = Kh + (size_t)(kv0 + j) * DKD + cb;
            f16* d = &Kl[j * 72 + cb];
#ifdef HAVE_ASYNC_LDS
            async_cp32B(d, s);
#else
            *(v8h*)(d)     = *(const v8h*)(s);
            *(v8h*)(d + 8) = *(const v8h*)(s + 8);
#endif
        }
        {   // V chunk -> LDS, transposed (so PV B-frags read contiguously)
            const int j = t & 63, db = (t >> 6) * 16;
            const f16* s = Vh + (size_t)(kv0 + j) * DKD + db;
            v8h a0 = *(const v8h*)(s), a1 = *(const v8h*)(s + 8);
            #pragma unroll
            for (int u = 0; u < 8; ++u) {
                Vt[(db + u)     * 72 + j] = a0[u];
                Vt[(db + 8 + u) * 72 + j] = a1[u];
            }
        }
#ifdef HAVE_ASYNC_LDS
        wait_async0();
#endif
        __syncthreads();

        if (kv0 + 64 < SEQ) {   // prefetch next chunk (global_prefetch_b8)
            __builtin_prefetch(Kh + (size_t)(kv0 + 64 + (t >> 2)) * DKD, 0, 1);
            __builtin_prefetch(Vh + (size_t)(kv0 + 64 + (t & 63)) * DKD, 0, 1);
        }

        // Scores: S = (Q K^T) * 1/8, then mask
        float sc[4][8];
        #pragma unroll
        for (int nt = 0; nt < 4; ++nt) {
            v8f s8 = zero8();
            #pragma unroll
            for (int ks = 0; ks < 2; ++ks) {
                v16h bfr = frag_b(&Kl[ks * 32], 72, nt * 16 + ln, hi);
                s8 = wmma_f16(qf[ks], bfr, s8);
            }
            #pragma unroll
            for (int r = 0; r < 8; ++r) {
                const int q   = qb + r + 8 * hi;
                const int kvc = kv0 + nt * 16 + ln;
                float sv = s8[r] * 0.125f;                 // 1/sqrt(64)
                if (mask[(size_t)q * SEQ + kvc] == 0) sv = -1.0e9f;
                sc[nt][r] = sv;
            }
        }

        // Online softmax; rows live in 16-lane groups (lanes 0-15 / 16-31)
        #pragma unroll
        for (int r = 0; r < 8; ++r) {
            float mx = fmaxf(fmaxf(sc[0][r], sc[1][r]), fmaxf(sc[2][r], sc[3][r]));
            #pragma unroll
            for (int d = 1; d < 16; d <<= 1) mx = fmaxf(mx, __shfl_xor(mx, d, 32));
            const float nm = fmaxf(M[r], mx);
            const float al = __expf(M[r] - nm);
            M[r] = nm;
            float rs = 0.0f;
            #pragma unroll
            for (int nt = 0; nt < 4; ++nt) {
                const float p = __expf(sc[nt][r] - nm);
                rs += p;
                Pw[(r + 8 * hi) * 72 + nt * 16 + ln] = (f16)p;
            }
            #pragma unroll
            for (int d = 1; d < 16; d <<= 1) rs += __shfl_xor(rs, d, 32);
            L[r] = L[r] * al + rs;
            #pragma unroll
            for (int nt = 0; nt < 4; ++nt) o[nt][r] *= al;
        }

        // Wave-local fence: P ds_store_b16 must land before A-frag ds reads
        asm volatile("s_wait_dscnt 0x0" ::: "memory");

        // O += P @ V
        #pragma unroll
        for (int nt = 0; nt < 4; ++nt) {
            #pragma unroll
            for (int ks = 0; ks < 2; ++ks) {
                v16h af = frag_a(&Pw[ks * 32], 72, ln, hi);
                v16h bf = frag_b(&Vt[ks * 32], 72, nt * 16 + ln, hi);
                o[nt] = wmma_f16(af, bf, o[nt]);
            }
        }
        __syncthreads();
    }

    // Normalize and write O as f16 in [B,S,D] (head-concatenated)
    #pragma unroll
    for (int nt = 0; nt < 4; ++nt) {
        const int col = h * DKD + nt * 16 + ln;
        #pragma unroll
        for (int r = 0; r < 8; ++r) {
            const int q = qb + r + 8 * hi;
            const float val = o[nt][r] / fmaxf(L[r], 1e-20f);
            O16[((size_t)b * SEQ + q) * BDIM + col] = (f16)val;
        }
    }
}

// ---------------------------------------------------------------------------
// Kernel 3: output projection.  out = O16 @ Wo^T + bo, fp32 output.
// ---------------------------------------------------------------------------
__global__ __launch_bounds__(256)
void out_proj_kernel(const f16* __restrict__ A, const float* __restrict__ W,
                     const float* __restrict__ bias, float* __restrict__ out)
{
    __shared__ f16 lA[128 * 40];
    __shared__ f16 lB[128 * 40];

    const int t    = threadIdx.x;
    const int lane = t & 31;
    const int w    = t >> 5;
    const int hi   = lane >> 4;
    const int ln   = lane & 15;
    const int wm   = (w & 3) * 32;
    const int wn   = (w >> 2) * 64;

    const int Mbase = blockIdx.y * 128;
    const int Nbase = blockIdx.x * 128;
    const int fr = t >> 1;
    const int fc = (t & 1) * 16;

    v8f acc[2][4];
    #pragma unroll
    for (int i = 0; i < 2; ++i)
        #pragma unroll
        for (int j = 0; j < 4; ++j) acc[i][j] = zero8();

    for (int k0 = 0; k0 < BDIM; k0 += 32) {
        {   // A tile: already f16 -> async copy when available
            const f16* s = A + (size_t)(Mbase + fr) * BDIM + k0 + fc;
            f16* d = &lA[fr * 40 + fc];
#ifdef HAVE_ASYNC_LDS
            async_cp32B(d, s);
#else
            *(v8h*)(d)     = *(const v8h*)(s);
            *(v8h*)(d + 8) = *(const v8h*)(s + 8);
#endif
        }
        {   // B tile: Wo fp32 -> f16
            v8h h0, h1;
            cvt16(W + (size_t)(Nbase + fr) * BDIM + k0 + fc, h0, h1);
            *(v8h*)(&lB[fr * 40 + fc])     = h0;
            *(v8h*)(&lB[fr * 40 + fc + 8]) = h1;
        }
#ifdef HAVE_ASYNC_LDS
        wait_async0();
#endif
        __syncthreads();

        v16h af[2], bf[4];
        #pragma unroll
        for (int mt = 0; mt < 2; ++mt)
            af[mt] = frag_a(&lA[(wm + mt * 16) * 40], 40, ln, hi);
        #pragma unroll
        for (int nt = 0; nt < 4; ++nt)
            bf[nt] = frag_b(&lB[(wn + nt * 16) * 40], 40, ln, hi);
        #pragma unroll
        for (int mt = 0; mt < 2; ++mt)
            #pragma unroll
            for (int nt = 0; nt < 4; ++nt)
                acc[mt][nt] = wmma_f16(af[mt], bf[nt], acc[mt][nt]);
        __syncthreads();
    }

    #pragma unroll
    for (int nt = 0; nt < 4; ++nt) {
        const int e  = Nbase + wn + nt * 16 + ln;
        const float be = bias[e];
        #pragma unroll
        for (int mt = 0; mt < 2; ++mt) {
            #pragma unroll
            for (int r = 0; r < 8; ++r) {
                const int i = Mbase + wm + mt * 16 + r + 8 * hi;
                out[(size_t)i * BDIM + e] = acc[mt][nt][r] + be;
            }
        }
    }
}

// ---------------------------------------------------------------------------
extern "C" void kernel_launch(void* const* d_in, const int* in_sizes, int n_in,
                              void* d_out, int out_size, void* d_ws, size_t ws_size,
                              hipStream_t stream) {
    (void)in_sizes; (void)n_in; (void)out_size; (void)ws_size;
    const float* query = (const float*)d_in[0];
    const float* key_  = (const float*)d_in[1];
    const float* value = (const float*)d_in[2];
    const int*   mask  = (const int*)d_in[3];
    const float* Wq = (const float*)d_in[4];
    const float* bq = (const float*)d_in[5];
    const float* Wk = (const float*)d_in[6];
    const float* bk = (const float*)d_in[7];
    const float* Wv = (const float*)d_in[8];
    const float* bv = (const float*)d_in[9];
    const float* Wo = (const float*)d_in[10];
    const float* bo = (const float*)d_in[11];
    float* out = (float*)d_out;

    const size_t elems = (size_t)NB * SEQ * BDIM;  // 8,388,608
    f16* Q16 = (f16*)d_ws;
    f16* K16 = Q16 + elems;
    f16* V16 = K16 + elems;
    f16* O16 = V16 + elems;

    qkv_proj_kernel<<<dim3(BDIM / 128, (NB * SEQ) / 128, 3), 256, 0, stream>>>(
        query, key_, value, Wq, bq, Wk, bk, Wv, bv, Q16, K16, V16);

    attention_kernel<<<dim3(SEQ / 128, NH, NB), 256, 0, stream>>>(
        Q16, K16, V16, mask, O16);

    out_proj_kernel<<<dim3(BDIM / 128, (NB * SEQ) / 128, 1), 256, 0, stream>>>(
        O16, Wo, bo, out);
}
